// LlamaAttention_67834713473571
// MI455X (gfx1250) — matrix operand order repaired and checked by reading, not compiled, not run on hardware
//
#include <hip/hip_runtime.h>
#include <hip/hip_bf16.h>
#include <stdint.h>

// ---------------------------------------------------------------- constants
#define S_LEN 2048
#define HID   4096
#define NH    32
#define NKV   8
#define HD    128
#define HALF  64

typedef __attribute__((ext_vector_type(4)))  __bf16 v4bf;
typedef __attribute__((ext_vector_type(8)))  __bf16 v8bf;
typedef __attribute__((ext_vector_type(16))) __bf16 v16bf;
typedef __attribute__((ext_vector_type(8)))  float  v8f;

// ---------------------------------------------------------------- fragment helpers
// A fragment (16x32 bf16, MxK): lane L holds row M=L%16; elements e<8 are
// K = k0 + 8*(L/16) + e, elements e>=8 are K = k0 + 16 + 8*(L/16) + (e-8).
// Both chunks are 8 contiguous bf16 (16B) -> two b128 loads.
__device__ __forceinline__ v16bf load_a_frag(const __bf16* base, int ld, int k0, int lane) {
    const __bf16* p = base + (size_t)(lane & 15) * ld + k0 + ((lane >> 4) << 3);
    v8bf lo = *(const v8bf*)p;
    v8bf hi = *(const v8bf*)(p + 16);
    v16bf f;
#pragma unroll
    for (int i = 0; i < 8; ++i) { f[i] = lo[i]; f[i + 8] = hi[i]; }
    return f;
}

// B fragment (32x16 bf16, KxN) for B^T stored row-major as [N, Kdim] with
// leading dim ld: lane L holds column N=n0+L%16, elements e<8 are
// K = k0 + 8*(L/16) + e, elements e>=8 are K = k0 + 16 + 8*(L/16) + (e-8).
__device__ __forceinline__ v16bf load_b_frag(const __bf16* base, int ld, int n0, int k0, int lane) {
    const __bf16* p = base + (size_t)(n0 + (lane & 15)) * ld + k0 + ((lane >> 4) << 3);
    v8bf lo = *(const v8bf*)p;
    v8bf hi = *(const v8bf*)(p + 16);
    v16bf f;
#pragma unroll
    for (int i = 0; i < 8; ++i) { f[i] = lo[i]; f[i + 8] = hi[i]; }
    return f;
}

__device__ __forceinline__ v8f wmma_bf16(v16bf a, v16bf b, v8f c) {
    return __builtin_amdgcn_wmma_f32_16x16x32_bf16(false, a, false, b, (short)0, c, false, false);
}

// ---------------------------------------------------------------- cast f32 -> bf16
__global__ __launch_bounds__(256) void cast_bf16_kernel(const float* __restrict__ src,
                                                        __bf16* __restrict__ dst, int n4) {
    int i = blockIdx.x * blockDim.x + threadIdx.x;
    if (i >= n4) return;
    float4 f = ((const float4*)src)[i];
    v4bf o;
    o[0] = (__bf16)f.x; o[1] = (__bf16)f.y; o[2] = (__bf16)f.z; o[3] = (__bf16)f.w;
    ((v4bf*)dst)[i] = o;
}

// ---------------------------------------------------------------- GEMM: C = X @ W^T
// X [M,K] bf16 row-major, W [N,K] bf16 row-major, C [M,N] f32.
// Block = 256 threads = 8 waves; wave tile 64x64; block tile 128x256.
// NT=true stores C with non-temporal hint (write-once output, keep L2 for weights).
template <bool NT>
__global__ __launch_bounds__(256) void gemm_xwt_kernel(const __bf16* __restrict__ X,
                                                       const __bf16* __restrict__ W,
                                                       float* __restrict__ C,
                                                       int N, int K) {
    const int lane = threadIdx.x & 31;
    const int wave = threadIdx.x >> 5;
    const int m0 = blockIdx.y * 128 + (wave & 1) * 64;
    const int n0 = blockIdx.x * 256 + (wave >> 1) * 64;

    v8f acc[4][4];
#pragma unroll
    for (int i = 0; i < 4; ++i)
#pragma unroll
        for (int j = 0; j < 4; ++j) acc[i][j] = (v8f)0.0f;

    const int arow = lane & 15;
    const int aoff = (lane >> 4) << 3;
    const int boff = (lane >> 4) << 4;

    for (int k0 = 0; k0 < K; k0 += 32) {
        // prefetch two k-steps ahead (compiler already pipelines one step) —
        // one byte per lane pulls the whole cacheline (global_prefetch_b8)
        if (k0 + 64 < K) {
#pragma unroll
            for (int i = 0; i < 4; ++i)
                __builtin_prefetch(X + (size_t)(m0 + 16 * i + arow) * K + (k0 + 64) + aoff, 0, 3);
#pragma unroll
            for (int j = 0; j < 4; ++j)
                __builtin_prefetch(W + (size_t)(n0 + 16 * j + arow) * K + (k0 + 64) + boff, 0, 3);
        }
        v16bf a[4], b[4];
#pragma unroll
        for (int i = 0; i < 4; ++i)
            a[i] = load_a_frag(X + (size_t)(m0 + 16 * i) * K, K, k0, lane);
#pragma unroll
        for (int j = 0; j < 4; ++j)
            b[j] = load_b_frag(W, K, n0 + 16 * j, k0, lane);
#pragma unroll
        for (int i = 0; i < 4; ++i)
#pragma unroll
            for (int j = 0; j < 4; ++j)
                acc[i][j] = wmma_bf16(a[i], b[j], acc[i][j]);
    }

    const int rlo = (lane >> 4) << 3;   // C layout: row = m0+16i + rlo + r
    const int cl  = lane & 15;          //           col = n0+16j + cl
#pragma unroll
    for (int i = 0; i < 4; ++i)
#pragma unroll
        for (int j = 0; j < 4; ++j)
#pragma unroll
            for (int r = 0; r < 8; ++r) {
                float* p = C + (size_t)(m0 + 16 * i + rlo + r) * N + (n0 + 16 * j + cl);
                if (NT) __builtin_nontemporal_store(acc[i][j][r], p);
                else    *p = acc[i][j][r];
            }
}

// ---------------------------------------------------------------- RoPE + layout [H,S,HD]
__global__ __launch_bounds__(256) void rope_arrange_kernel(const float* __restrict__ src,
                                                           __bf16* __restrict__ dst,
                                                           const float* __restrict__ cosp,
                                                           const float* __restrict__ sinp,
                                                           int H) {
    int idx = blockIdx.x * blockDim.x + threadIdx.x;
    int total = S_LEN * H * HALF;
    if (idx >= total) return;
    int d = idx & (HALF - 1);
    int t = idx >> 6;
    int h = t % H;
    int s = t / H;
    const float* row = src + (size_t)s * (H * HD) + h * HD;
    float x1 = row[d];
    float x2 = row[HALF + d];
    float c  = cosp[s * HALF + d];
    float sn = sinp[s * HALF + d];
    __bf16* orow = dst + ((size_t)h * S_LEN + s) * HD;
    orow[d]        = (__bf16)(x1 * c - x2 * sn);
    orow[HALF + d] = (__bf16)(x2 * c + x1 * sn);
}

// ---------------------------------------------------------------- V transpose -> [KV, HD, S]
__global__ __launch_bounds__(256) void transpose_v_kernel(const float* __restrict__ v,
                                                          __bf16* __restrict__ vt) {
    int idx = blockIdx.x * blockDim.x + threadIdx.x;
    if (idx >= NKV * HD * S_LEN) return;
    int s = idx & (S_LEN - 1);
    int c = idx >> 11;          // c = kv*HD + d
    vt[idx] = (__bf16)(v[(size_t)s * (NKV * HD) + c]);
}

// ---------------------------------------------------------------- flash attention
// Q [NH,S,HD] bf16, K [NKV,S,HD] bf16, Vt [NKV,HD,S] bf16, O [S,NH*HD] bf16.
// Block = 128 threads = 4 waves, wave owns 16 query rows; online softmax.
__global__ __launch_bounds__(128) void flash_attn_kernel(const __bf16* __restrict__ Q,
                                                         const __bf16* __restrict__ Kc,
                                                         const __bf16* __restrict__ Vt,
                                                         __bf16* __restrict__ O) {
    __shared__ __bf16 plds[4][16][32];   // per-wave P tile (16 rows x 32 keys)
    const int lane = threadIdx.x & 31;
    const int w    = threadIdx.x >> 5;
    const int h    = blockIdx.y;
    const int kv   = h & (NKV - 1);      // jnp.tile semantics: kv head = h % NKV
    const int q0   = blockIdx.x * 64 + w * 16;

    const __bf16* qb = Q  + ((size_t)h  * S_LEN + q0) * HD;
    const __bf16* kb = Kc + (size_t)kv * S_LEN * HD;
    const __bf16* vb = Vt + (size_t)kv * HD * S_LEN;

    v16bf aq[4];
#pragma unroll
    for (int d = 0; d < 4; ++d) aq[d] = load_a_frag(qb, HD, d * 32, lane);

    v8f acc[8];
#pragma unroll
    for (int t = 0; t < 8; ++t) acc[t] = (v8f)0.0f;
    float mrow[8], lrow[8];
#pragma unroll
    for (int r = 0; r < 8; ++r) { mrow[r] = -1e30f; lrow[r] = 0.0f; }

    const float scale = 0.08838834764831845f;   // 1/sqrt(128)
    const int rbase = q0 + ((lane >> 4) << 3);
    const int cbase = lane & 15;

    for (int kv0 = 0; kv0 < q0 + 16; kv0 += 32) {
        // scores: two 16x16 tiles over K=HD=128
        v8f s0 = (v8f)0.0f, s1 = (v8f)0.0f;
#pragma unroll
        for (int d = 0; d < 4; ++d) {
            v16bf b0 = load_b_frag(kb, HD, kv0,      d * 32, lane);
            v16bf b1 = load_b_frag(kb, HD, kv0 + 16, d * 32, lane);
            s0 = wmma_bf16(aq[d], b0, s0);
            s1 = wmma_bf16(aq[d], b1, s1);
        }
        // online softmax per row; write P tile to LDS in A-fragment-friendly form
#pragma unroll
        for (int r = 0; r < 8; ++r) {
            int row = rbase + r;
            float e0 = (kv0 + cbase      > row) ? -1e30f : s0[r] * scale;
            float e1 = (kv0 + 16 + cbase > row) ? -1e30f : s1[r] * scale;
            float mx = fmaxf(e0, e1);
#pragma unroll
            for (int o = 1; o < 16; o <<= 1) mx = fmaxf(mx, __shfl_xor(mx, o, 32));
            float mnew = fmaxf(mrow[r], mx);
            float corr = expf(mrow[r] - mnew);
            float p0 = expf(e0 - mnew);
            float p1 = expf(e1 - mnew);
            float ps = p0 + p1;
#pragma unroll
            for (int o = 1; o < 16; o <<= 1) ps += __shfl_xor(ps, o, 32);
            lrow[r] = lrow[r] * corr + ps;
            mrow[r] = mnew;
#pragma unroll
            for (int t = 0; t < 8; ++t) acc[t][r] *= corr;
            int M = ((lane >> 4) << 3) + r;
            plds[w][M][cbase]      = (__bf16)p0;
            plds[w][M][16 + cbase] = (__bf16)p1;
        }
        asm volatile("" ::: "memory");   // keep ds stores before ds loads (LDS is in-order per wave)
        // reload P as a 16x32 A fragment
        const __bf16* rp = &plds[w][lane & 15][(lane >> 4) << 3];
        v8bf lo = *(const v8bf*)rp;
        v8bf hi = *(const v8bf*)(rp + 16);
        v16bf pa;
#pragma unroll
        for (int i = 0; i < 8; ++i) { pa[i] = lo[i]; pa[i + 8] = hi[i]; }
        // out += P @ V  (Vt rows are contiguous in key dim)
#pragma unroll
        for (int t = 0; t < 8; ++t) {
            v16bf bv = load_b_frag(vb, S_LEN, t * 16, kv0, lane);
            acc[t] = wmma_bf16(pa, bv, acc[t]);
        }
        asm volatile("" ::: "memory");
    }

#pragma unroll
    for (int r = 0; r < 8; ++r) {
        float inv = 1.0f / lrow[r];
        size_t row = (size_t)(rbase + r);
#pragma unroll
        for (int t = 0; t < 8; ++t)
            O[row * (NH * HD) + (size_t)h * HD + t * 16 + cbase] = (__bf16)(acc[t][r] * inv);
    }
}

// ---------------------------------------------------------------- launch
extern "C" void kernel_launch(void* const* d_in, const int* in_sizes, int n_in,
                              void* d_out, int out_size, void* d_ws, size_t ws_size,
                              hipStream_t stream) {
    const float* x    = (const float*)d_in[0];   // [1,S,HID]
    const float* wq   = (const float*)d_in[1];   // [HID, HID]
    const float* wk   = (const float*)d_in[2];   // [NKV*HD, HID]
    const float* wv   = (const float*)d_in[3];   // [NKV*HD, HID]
    const float* wo   = (const float*)d_in[4];   // [HID, HID]
    const float* cosp = (const float*)d_in[5];   // [1,1,S,HALF]
    const float* sinp = (const float*)d_in[6];   // [1,1,S,HALF]
    float* out = (float*)d_out;                  // [1,S,HID]

    char* ws = (char*)d_ws;
    size_t off = 0;
    __bf16* xbf  = (__bf16*)(ws + off); off += (size_t)S_LEN * HID * 2;          // 16 MB
    __bf16* wqbf = (__bf16*)(ws + off); off += (size_t)HID * HID * 2;            // 32 MB
    __bf16* wkbf = (__bf16*)(ws + off); off += (size_t)NKV * HD * HID * 2;       //  8 MB
    __bf16* wvbf = (__bf16*)(ws + off); off += (size_t)NKV * HD * HID * 2;       //  8 MB
    __bf16* wobf = (__bf16*)(ws + off); off += (size_t)HID * HID * 2;            // 32 MB
    float*  qf   = (float*)(ws + off);  off += (size_t)S_LEN * HID * 4;          // 32 MB
    float*  kf   = (float*)(ws + off);  off += (size_t)S_LEN * NKV * HD * 4;     //  8 MB
    float*  vf   = (float*)(ws + off);  off += (size_t)S_LEN * NKV * HD * 4;     //  8 MB
    __bf16* qbf  = (__bf16*)(ws + off); off += (size_t)NH * S_LEN * HD * 2;      // 16 MB
    __bf16* kbf  = (__bf16*)(ws + off); off += (size_t)NKV * S_LEN * HD * 2;     //  4 MB
    __bf16* vtbf = (__bf16*)(ws + off); off += (size_t)NKV * HD * S_LEN * 2;     //  4 MB
    __bf16* abf  = (__bf16*)(ws + off); off += (size_t)S_LEN * HID * 2;          // 16 MB

    dim3 blk(256);

    // 1) casts to bf16
    cast_bf16_kernel<<<(S_LEN * HID / 4 + 255) / 256, blk, 0, stream>>>(x,  xbf,  S_LEN * HID / 4);
    cast_bf16_kernel<<<(HID * HID / 4 + 255) / 256, blk, 0, stream>>>(wq, wqbf, HID * HID / 4);
    cast_bf16_kernel<<<(NKV * HD * HID / 4 + 255) / 256, blk, 0, stream>>>(wk, wkbf, NKV * HD * HID / 4);
    cast_bf16_kernel<<<(NKV * HD * HID / 4 + 255) / 256, blk, 0, stream>>>(wv, wvbf, NKV * HD * HID / 4);
    cast_bf16_kernel<<<(HID * HID / 4 + 255) / 256, blk, 0, stream>>>(wo, wobf, HID * HID / 4);

    // 2) projections (C = X @ W^T), grid = (N/256, M/128)
    gemm_xwt_kernel<false><<<dim3(HID / 256, S_LEN / 128), blk, 0, stream>>>(xbf, wqbf, qf, HID, HID);
    gemm_xwt_kernel<false><<<dim3(NKV * HD / 256, S_LEN / 128), blk, 0, stream>>>(xbf, wkbf, kf, NKV * HD, HID);
    gemm_xwt_kernel<false><<<dim3(NKV * HD / 256, S_LEN / 128), blk, 0, stream>>>(xbf, wvbf, vf, NKV * HD, HID);

    // 3) RoPE + layout, V transpose
    rope_arrange_kernel<<<(S_LEN * NH * HALF + 255) / 256, blk, 0, stream>>>(qf, qbf, cosp, sinp, NH);
    rope_arrange_kernel<<<(S_LEN * NKV * HALF + 255) / 256, blk, 0, stream>>>(kf, kbf, cosp, sinp, NKV);
    transpose_v_kernel<<<(NKV * HD * S_LEN + 255) / 256, blk, 0, stream>>>(vf, vtbf);

    // 4) attention
    flash_attn_kernel<<<dim3(S_LEN / 64, NH), dim3(128), 0, stream>>>(qbf, kbf, vtbf, abf);

    // 5) output projection -> f32 d_out (non-temporal: write-once, keep L2 for weights)
    gemm_xwt_kernel<true><<<dim3(HID / 256, S_LEN / 128), blk, 0, stream>>>(abf, wobf, out, HID, HID);

    (void)in_sizes; (void)n_in; (void)out_size; (void)ws_size;
}
